// Decoder_32753420600133
// MI455X (gfx1250) — matrix operand order repaired and checked by reading, not compiled, not run on hardware
//
#include <hip/hip_runtime.h>
#include <hip/hip_bf16.h>
#include <math.h>

// ---------------------------------------------------------------------------
// Problem dims
// ---------------------------------------------------------------------------
#define BB   128
#define SS   512
#define TT   20
#define EE   128
#define HH   256
#define VV   50000
#define NO_  50
#define DD   512           // 2*H
#define VNO  (VV + NO_)    // 50050

typedef __attribute__((ext_vector_type(16))) __bf16 v16bf;
typedef __attribute__((ext_vector_type(8)))  __bf16 v8bf;
typedef __attribute__((ext_vector_type(4)))  __bf16 v4bf;
typedef __attribute__((ext_vector_type(8)))  float  v8f;

__device__ __forceinline__ __bf16 f2bf(float f) {
    union { float f; unsigned u; } x; x.f = f;
    unsigned r = (x.u + 0x7FFFu + ((x.u >> 16) & 1u)) >> 16;   // RNE
    union { unsigned short s; __bf16 b; } y; y.s = (unsigned short)r;
    return y.b;
}
__device__ __forceinline__ v4bf f2bf4(float4 v) {
    v4bf o; o[0] = f2bf(v.x); o[1] = f2bf(v.y); o[2] = f2bf(v.z); o[3] = f2bf(v.w);
    return o;
}
__device__ __forceinline__ float sigm(float x) { return 1.0f / (1.0f + __expf(-x)); }

// ---------------------------------------------------------------------------
// Generic WMMA GEMM:  out[M=128*gridY rows, N] = concat(A1,A2) @ concat(W1;W2) + bias
//   A split along K at AK1 (A2 may be null), W split along K at WK1 (W2 may be null)
//   Block: 256 threads (8 waves); block tile = 128 rows x 64 cols.
//   Requirements: K, AK1, WK1 multiples of 32; lda/ldw/N multiples of 4.
// ---------------------------------------------------------------------------
__global__ __launch_bounds__(256)
void gemm_bias_kernel(const float* __restrict__ A1, int lda1, int AK1,
                      const float* __restrict__ A2, int lda2,
                      const float* __restrict__ W1, int ldw1, int WK1,
                      const float* __restrict__ W2, int ldw2,
                      const float* __restrict__ bias,
                      float* __restrict__ out, int ldo,
                      int N, int K)
{
    __shared__ __bf16 lds_a[128 * 40];   // 128 rows x 32 k, stride 40 (16B aligned)
    __shared__ __bf16 lds_b[64 * 40];    // 64 cols  x 32 k (transposed), stride 40

    const int tid  = threadIdx.x;
    const int lane = tid & 31;
    const int wv   = tid >> 5;                 // wave id -> row tile
    const int rowBase = blockIdx.y * 128;
    const int n0      = blockIdx.x * 64;

    v8f acc[4];
    #pragma unroll
    for (int nt = 0; nt < 4; ++nt)
        #pragma unroll
        for (int e = 0; e < 8; ++e) acc[nt][e] = 0.0f;

    const int am = lane & 15, kh = lane >> 4, kb = kh * 16;

    for (int k0 = 0; k0 < K; k0 += 32) {
        // stage A tile 128x32: 1024 float4 loads (fp32 -> bf16, ds_store_b64)
        for (int e4 = tid; e4 < 1024; e4 += 256) {
            int r  = e4 >> 3;            // 8 float4 per row
            int kk = (e4 & 7) << 2;
            int k  = k0 + kk;
            float4 v = (k < AK1)
                ? *(const float4*)&A1[(size_t)(rowBase + r) * lda1 + k]
                : *(const float4*)&A2[(size_t)(rowBase + r) * lda2 + (k - AK1)];
            *(v4bf*)&lds_a[r * 40 + kk] = f2bf4(v);
        }
        // stage B tile 32x64 transposed -> lds_b[n][k]: 512 float4 loads
        for (int e4 = tid; e4 < 512; e4 += 256) {
            int k = e4 >> 4;             // 16 float4 per k-row
            int n = (e4 & 15) << 2;
            int gk = k0 + k, gn = n0 + n;
            float4 v = make_float4(0.f, 0.f, 0.f, 0.f);
            if (gn < N)                  // N % 4 == 0 -> whole float4 valid
                v = (gk < WK1) ? *(const float4*)&W1[(size_t)gk * ldw1 + gn]
                               : *(const float4*)&W2[(size_t)(gk - WK1) * ldw2 + gn];
            lds_b[(n + 0) * 40 + k] = f2bf(v.x);
            lds_b[(n + 1) * 40 + k] = f2bf(v.y);
            lds_b[(n + 2) * 40 + k] = f2bf(v.z);
            lds_b[(n + 3) * 40 + k] = f2bf(v.w);
            // prefetch next K-step weight tile
            int pk = gk + 32;
            if (pk < K && gn < N) {
                const float* p = (pk < WK1) ? &W1[(size_t)pk * ldw1 + gn]
                                            : &W2[(size_t)(pk - WK1) * ldw2 + gn];
                __builtin_prefetch(p, 0, 1);
            }
        }
        __syncthreads();

        // A fragment: lane=M, K chunks kh*8 and 16+kh*8
        v8bf alo = *(const v8bf*)&lds_a[(wv * 16 + am) * 40 + kh * 8];
        v8bf ahi = *(const v8bf*)&lds_a[(wv * 16 + am) * 40 + 16 + kh * 8];
        v16bf af = __builtin_shufflevector(alo, ahi, 0,1,2,3,4,5,6,7,8,9,10,11,12,13,14,15);

        #pragma unroll
        for (int nt = 0; nt < 4; ++nt) {
            // B fragment: lane=N, K chunks kh*8 and 16+kh*8
            v8bf b0 = *(const v8bf*)&lds_b[(nt * 16 + am) * 40 + kh * 8];
            v8bf b1 = *(const v8bf*)&lds_b[(nt * 16 + am) * 40 + 16 + kh * 8];
            v16bf bf = __builtin_shufflevector(b0, b1, 0,1,2,3,4,5,6,7,8,9,10,11,12,13,14,15);
            acc[nt] = __builtin_amdgcn_wmma_f32_16x16x32_bf16(
                false, af, false, bf, (short)0, acc[nt], false, false);
        }
        __syncthreads();
    }

    // epilogue: lane holds col n = lane&15, rows (lane>>4)*8 + e
    const int rowHalf = (lane >> 4) * 8;
    #pragma unroll
    for (int nt = 0; nt < 4; ++nt) {
        int n = n0 + nt * 16 + am;
        if (n < N) {
            float bv = bias ? bias[n] : 0.0f;
            #pragma unroll
            for (int e = 0; e < 8; ++e) {
                int row = rowBase + wv * 16 + rowHalf + e;
                out[(size_t)row * ldo + n] = acc[nt][e] + bv;
            }
        }
    }
}

// ---------------------------------------------------------------------------
// Fused attention scores:  out[row] = sum_n tanh( (A@W)[row,n] + bias[row/grp][n] ) * v[n]
//   A block (128 x K) staged in LDS once (K = 512 or 256, power of two).
// ---------------------------------------------------------------------------
__global__ __launch_bounds__(256)
void attn_score_kernel(const float* __restrict__ A, int lda, int kShift,
                       const float* __restrict__ W, int ldw,
                       const float* __restrict__ bias, int groupRows,
                       const float* __restrict__ v,
                       float* __restrict__ out,
                       int N, int K)
{
    extern __shared__ __bf16 smem[];
    const int KP = K + 8;                          // pad keeps 16B alignment
    __bf16* lds_a = smem;                          // 128 x KP
    __bf16* lds_b = smem + 128 * KP;               // 64 x 40

    const int tid  = threadIdx.x;
    const int lane = tid & 31;
    const int wv   = tid >> 5;
    const int rowBase = blockIdx.x * 128;

    // stage the whole A block once (float4 loads, enc_out read exactly once)
    const int kvecMask = (K >> 2) - 1;
    for (int e4 = tid; e4 < (128 * K) >> 2; e4 += 256) {
        int r = e4 >> (kShift - 2);
        int k = (e4 & kvecMask) << 2;
        float4 vv = *(const float4*)&A[(size_t)(rowBase + r) * lda + k];
        *(v4bf*)&lds_a[r * KP + k] = f2bf4(vv);
    }
    __syncthreads();

    float rowsum[8];
    const float* biasRow[8];
    const int am = lane & 15, kh = lane >> 4, kb = kh * 16;
    const int rowHalf = (lane >> 4) * 8;
    #pragma unroll
    for (int e = 0; e < 8; ++e) {
        rowsum[e] = 0.0f;
        int row = rowBase + wv * 16 + rowHalf + e;
        biasRow[e] = bias + (size_t)(row / groupRows) * N;
    }

    for (int n0 = 0; n0 < N; n0 += 64) {
        v8f acc[4];
        #pragma unroll
        for (int nt = 0; nt < 4; ++nt)
            #pragma unroll
            for (int e = 0; e < 8; ++e) acc[nt][e] = 0.0f;

        for (int k0 = 0; k0 < K; k0 += 32) {
            __syncthreads();   // protect lds_b reuse vs previous iteration's reads
            for (int e4 = tid; e4 < 512; e4 += 256) {
                int k = e4 >> 4;
                int n = (e4 & 15) << 2;
                float4 vv = *(const float4*)&W[(size_t)(k0 + k) * ldw + n0 + n];
                lds_b[(n + 0) * 40 + k] = f2bf(vv.x);
                lds_b[(n + 1) * 40 + k] = f2bf(vv.y);
                lds_b[(n + 2) * 40 + k] = f2bf(vv.z);
                lds_b[(n + 3) * 40 + k] = f2bf(vv.w);
                int pk = k0 + 32 + k;
                if (pk < K)
                    __builtin_prefetch(&W[(size_t)pk * ldw + n0 + n], 0, 1);
            }
            __syncthreads();

            v8bf alo = *(const v8bf*)&lds_a[(wv * 16 + am) * KP + k0 + kh * 8];
            v8bf ahi = *(const v8bf*)&lds_a[(wv * 16 + am) * KP + k0 + 16 + kh * 8];
            v16bf af = __builtin_shufflevector(alo, ahi, 0,1,2,3,4,5,6,7,8,9,10,11,12,13,14,15);

            #pragma unroll
            for (int nt = 0; nt < 4; ++nt) {
                v8bf b0 = *(const v8bf*)&lds_b[(nt * 16 + am) * 40 + kh * 8];
                v8bf b1 = *(const v8bf*)&lds_b[(nt * 16 + am) * 40 + 16 + kh * 8];
                v16bf bf = __builtin_shufflevector(b0, b1, 0,1,2,3,4,5,6,7,8,9,10,11,12,13,14,15);
                acc[nt] = __builtin_amdgcn_wmma_f32_16x16x32_bf16(
                    false, af, false, bf, (short)0, acc[nt], false, false);
            }
        }
        // fused tanh + dot(v) on the C fragments (never materialize [M,N])
        #pragma unroll
        for (int nt = 0; nt < 4; ++nt) {
            int n = n0 + nt * 16 + am;
            float vn = v[n];
            #pragma unroll
            for (int e = 0; e < 8; ++e) {
                float t = tanhf(acc[nt][e] + biasRow[e][n]);
                rowsum[e] += t * vn;
            }
        }
    }
    // reduce the 16 lanes of each half-wave (each half holds distinct rows)
    #pragma unroll
    for (int e = 0; e < 8; ++e) {
        float s = rowsum[e];
        s += __shfl_xor(s, 1, 32);
        s += __shfl_xor(s, 2, 32);
        s += __shfl_xor(s, 4, 32);
        s += __shfl_xor(s, 8, 32);
        rowsum[e] = s;
    }
    if ((lane & 15) == 0) {
        #pragma unroll
        for (int e = 0; e < 8; ++e)
            out[rowBase + wv * 16 + rowHalf + e] = rowsum[e];
    }
}

// ---------------------------------------------------------------------------
// Elementwise / reduction kernels
// ---------------------------------------------------------------------------
__global__ void lstm_kernel(const float* __restrict__ gates,
                            const float* __restrict__ s_c,
                            float* __restrict__ h_out, float* __restrict__ c_out,
                            float* __restrict__ st_hat)
{
    int b = blockIdx.x, t = threadIdx.x;                 // t = 0..255
    const float* g = gates + (size_t)b * 1024;
    float gi = g[t], gf = g[256 + t], gg = g[512 + t], go = g[768 + t];
    float c = sigm(gf) * s_c[b * HH + t] + sigm(gi) * tanhf(gg);
    float h = sigm(go) * tanhf(c);
    h_out[b * HH + t] = h;
    c_out[b * HH + t] = c;
    st_hat[b * DD + t] = h;
    st_hat[b * DD + HH + t] = c;
}

__global__ void prevs_copy_kernel(const float* __restrict__ prev_s,
                                  const float* __restrict__ h,
                                  float* __restrict__ out)   // [B, T+1, H]
{
    int i = blockIdx.x * 256 + threadIdx.x;               // < B*(T+1)*H
    int hh = i % HH;
    int bt = i / HH;
    int t = bt % (TT + 1);
    int b = bt / (TT + 1);
    out[i] = (t < TT) ? prev_s[((size_t)b * TT + t) * HH + hh] : h[b * HH + hh];
}

__global__ void attn_e_finish_kernel(const float* __restrict__ et,
                                     const float* __restrict__ sum_temporal,
                                     const float* __restrict__ mask,
                                     const float* __restrict__ enc_out,
                                     float* __restrict__ sum_new,
                                     float* __restrict__ at_out,
                                     float* __restrict__ ct_e_out)
{
    __shared__ float at_s[SS];
    __shared__ float red[256];
    int b = blockIdx.x, tid = threadIdx.x;
    float local = 0.0f;
    for (int s = tid; s < SS; s += 256) {
        float ex = __expf(et[b * SS + s]);
        float st = sum_temporal[b * SS + s];
        sum_new[b * SS + s] = st + ex;
        float a = (ex / st) * mask[b * SS + s];
        at_s[s] = a;
        local += a;
    }
    red[tid] = local; __syncthreads();
    for (int s = 128; s > 0; s >>= 1) { if (tid < s) red[tid] += red[tid + s]; __syncthreads(); }
    float inv = 1.0f / red[0];
    __syncthreads();
    for (int s = tid; s < SS; s += 256) {
        at_s[s] *= inv;
        at_out[b * SS + s] = at_s[s];
    }
    __syncthreads();
    // context vector: one float2 per thread (DD = 512 = 2*256)
    const float2* eo = (const float2*)(enc_out + (size_t)b * SS * DD) + tid;
    float ax = 0.0f, ay = 0.0f;
    for (int s = 0; s < SS; ++s) {
        float2 vv = eo[(size_t)s * (DD / 2)];
        float a = at_s[s];
        ax += a * vv.x;
        ay += a * vv.y;
    }
    float2 r; r.x = ax; r.y = ay;
    ((float2*)(ct_e_out + (size_t)b * DD))[tid] = r;
}

__global__ void attn_d_finish_kernel(const float* __restrict__ ed,
                                     const float* __restrict__ prev_s,
                                     float* __restrict__ ct_d)
{
    __shared__ float w[TT];
    int b = blockIdx.x, tid = threadIdx.x;
    if (tid == 0) {
        float mx = -1e30f;
        for (int t = 0; t < TT; ++t) mx = fmaxf(mx, ed[b * TT + t]);
        float sm = 0.0f;
        for (int t = 0; t < TT; ++t) { w[t] = __expf(ed[b * TT + t] - mx); sm += w[t]; }
        float inv = 1.0f / sm;
        for (int t = 0; t < TT; ++t) w[t] *= inv;
    }
    __syncthreads();
    float acc = 0.0f;
    for (int t = 0; t < TT; ++t) acc += w[t] * prev_s[((size_t)b * TT + t) * HH + tid];
    ct_d[b * HH + tid] = acc;
}

__global__ void pgen_kernel(const float* __restrict__ ct_e,
                            const float* __restrict__ ct_d,
                            const float* __restrict__ st_hat,
                            const float* __restrict__ x,
                            const float* __restrict__ W_pg,
                            const float* __restrict__ b_pg,
                            float* __restrict__ p_gen)
{
    __shared__ float red[256];
    int b = blockIdx.x, tid = threadIdx.x;
    float acc = 0.0f;
    for (int j = tid; j < 1408; j += 256) {
        float v;
        if      (j < 512)  v = ct_e[b * DD + j];
        else if (j < 768)  v = ct_d[b * HH + (j - 512)];
        else if (j < 1280) v = st_hat[b * DD + (j - 768)];
        else               v = x[b * EE + (j - 1280)];
        acc += v * W_pg[j];
    }
    red[tid] = acc; __syncthreads();
    for (int s = 128; s > 0; s >>= 1) { if (tid < s) red[tid] += red[tid + s]; __syncthreads(); }
    if (tid == 0) p_gen[b] = sigm(red[0] + b_pg[0]);
}

__global__ void concat_z_kernel(const float* __restrict__ h,
                                const float* __restrict__ ct_e,
                                const float* __restrict__ ct_d,
                                float* __restrict__ zin)   // [B, 1024]
{
    int i = blockIdx.x * 256 + threadIdx.x;               // < B*1024
    int j = i & 1023, b = i >> 10;
    float v;
    if      (j < 256) v = h[b * HH + j];
    else if (j < 768) v = ct_e[b * DD + (j - 256)];
    else              v = ct_d[b * HH + (j - 768)];
    zin[i] = v;
}

__global__ void softmax_stats_kernel(const float* __restrict__ logits,
                                     float* __restrict__ row_max,
                                     float* __restrict__ row_sum)
{
    __shared__ float red[256];
    int b = blockIdx.x, tid = threadIdx.x;
    float mx = -1e30f;
    for (int v = tid; v < VV; v += 256) mx = fmaxf(mx, logits[(size_t)b * VV + v]);
    red[tid] = mx; __syncthreads();
    for (int s = 128; s > 0; s >>= 1) { if (tid < s) red[tid] = fmaxf(red[tid], red[tid + s]); __syncthreads(); }
    mx = red[0]; __syncthreads();
    float sm = 0.0f;
    for (int v = tid; v < VV; v += 256) sm += __expf(logits[(size_t)b * VV + v] - mx);
    red[tid] = sm; __syncthreads();
    for (int s = 128; s > 0; s >>= 1) { if (tid < s) red[tid] += red[tid + s]; __syncthreads(); }
    if (tid == 0) { row_max[b] = mx; row_sum[b] = red[0]; }
}

__global__ void final_write_kernel(const float* __restrict__ logits,
                                   const float* __restrict__ row_max,
                                   const float* __restrict__ row_sum,
                                   const float* __restrict__ p_gen,
                                   float* __restrict__ final)
{
    int col = blockIdx.x * 256 + threadIdx.x;
    int b = blockIdx.y;
    if (col >= VNO) return;
    float v = 0.0f;
    if (col < VV)
        v = p_gen[b] * __expf(logits[(size_t)b * VV + col] - row_max[b]) / row_sum[b];
    final[(size_t)b * VNO + col] = v;
}

__global__ void scatter_kernel(const int* __restrict__ idx,
                               const float* __restrict__ at,
                               const float* __restrict__ p_gen,
                               float* __restrict__ final)
{
    int i = blockIdx.x * 256 + threadIdx.x;               // < B*S
    int b = i >> 9;
    float w = (1.0f - p_gen[b]) * at[i];
    atomicAdd(&final[(size_t)b * VNO + idx[i]], w);
}

// ---------------------------------------------------------------------------
// Launch
// ---------------------------------------------------------------------------
extern "C" void kernel_launch(void* const* d_in, const int* in_sizes, int n_in,
                              void* d_out, int out_size, void* d_ws, size_t ws_size,
                              hipStream_t stream)
{
    const float* x_t     = (const float*)d_in[0];
    const float* s_h     = (const float*)d_in[1];
    const float* s_c     = (const float*)d_in[2];
    const float* enc_out = (const float*)d_in[3];
    const float* mask    = (const float*)d_in[4];
    const float* ct_e    = (const float*)d_in[5];
    const float* sum_t   = (const float*)d_in[7];
    const float* prev_s  = (const float*)d_in[8];
    const float* W_xc    = (const float*)d_in[9];
    const float* b_xc    = (const float*)d_in[10];
    const float* W_ih    = (const float*)d_in[11];
    const float* W_hh    = (const float*)d_in[12];
    const float* b_lstm  = (const float*)d_in[13];
    const float* We_h    = (const float*)d_in[14];
    const float* We_s    = (const float*)d_in[15];
    const float* be_s    = (const float*)d_in[16];
    const float* ve      = (const float*)d_in[17];
    const float* Wd_prev = (const float*)d_in[18];
    const float* Wd_s    = (const float*)d_in[19];
    const float* bd_s    = (const float*)d_in[20];
    const float* vd      = (const float*)d_in[21];
    const float* W_pg    = (const float*)d_in[22];
    const float* b_pg    = (const float*)d_in[23];
    const float* W_V     = (const float*)d_in[24];
    const float* b_V     = (const float*)d_in[25];
    const float* W_V1    = (const float*)d_in[26];
    const float* b_V1    = (const float*)d_in[27];
    const int*   evocab  = (const int*)d_in[28];

    float* out = (float*)d_out;
    float* out_final = out;                                   // [B, V+NO]
    float* out_h     = out + (size_t)BB * VNO;                // [B, H]
    float* out_c     = out_h + (size_t)BB * HH;               // [B, H]
    float* out_cte   = out_c + (size_t)BB * HH;               // [B, D]
    float* out_sum   = out_cte + (size_t)BB * DD;             // [B, S]
    float* out_prevs = out_sum + (size_t)BB * SS;             // [B, T+1, H]

    float* ws = (float*)d_ws;
    float* ws_x     = ws;                     // [B, E]
    float* ws_st    = ws_x    + BB * EE;      // [B, D]
    float* ws_gates = ws_st   + BB * DD;      // [B, 4H]
    float* ws_proje = ws_gates+ BB * 4 * HH;  // [B, D]
    float* ws_et    = ws_proje+ BB * DD;      // [B, S]
    float* ws_at    = ws_et   + BB * SS;      // [B, S]
    float* ws_projd = ws_at   + BB * SS;      // [B, H]
    float* ws_ed    = ws_projd+ BB * HH;      // [B, T]
    float* ws_ctd   = ws_ed   + BB * TT;      // [B, H]
    float* ws_pg    = ws_ctd  + BB * HH;      // [B]
    float* ws_zin   = ws_pg   + BB;           // [B, 4H]
    float* ws_z     = ws_zin  + BB * 4 * HH;  // [B, H]
    float* ws_logit = ws_z    + BB * HH;      // [B, V]
    float* ws_rmax  = ws_logit+ (size_t)BB * VV;  // [B]
    float* ws_rsum  = ws_rmax + BB;               // [B]

    // 1) x = [x_t | ct_e] @ W_xc + b_xc            (M=128, N=128, K=640)
    gemm_bias_kernel<<<dim3(2, 1), 256, 0, stream>>>(
        x_t, EE, EE, ct_e, DD, W_xc, EE, 640, nullptr, 0, b_xc, ws_x, EE, EE, 640);

    // 2) gates = [x | s_h] @ [W_ih ; W_hh] + b     (M=128, N=1024, K=384)
    gemm_bias_kernel<<<dim3(16, 1), 256, 0, stream>>>(
        ws_x, EE, EE, s_h, HH, W_ih, 4 * HH, EE, W_hh, 4 * HH, b_lstm,
        ws_gates, 4 * HH, 4 * HH, 384);

    // 3) LSTM cell -> h, c, st_hat
    lstm_kernel<<<BB, 256, 0, stream>>>(ws_gates, s_c, out_h, out_c, ws_st);

    // 4) prev_s_new = concat(prev_s, h)
    prevs_copy_kernel<<<(BB * (TT + 1) * HH) / 256, 256, 0, stream>>>(prev_s, out_h, out_prevs);

    // 5) proj_e = st_hat @ We_s + be_s             (M=128, N=512, K=512)
    gemm_bias_kernel<<<dim3(8, 1), 256, 0, stream>>>(
        ws_st, DD, DD, nullptr, 0, We_s, DD, DD, nullptr, 0, be_s, ws_proje, DD, DD, DD);

    // 6) et = tanh(enc_out @ We_h + proj_e) . ve   (M=65536, N=512, K=512)
    {
        size_t shmem = (size_t)(128 * (DD + 8) + 64 * 40) * sizeof(__bf16);
        attn_score_kernel<<<(BB * SS) / 128, 256, shmem, stream>>>(
            enc_out, DD, 9, We_h, DD, ws_proje, SS, ve, ws_et, DD, DD);
    }

    // 7) temporal attention finish: sum_new, at, ct_e_new
    attn_e_finish_kernel<<<BB, 256, 0, stream>>>(
        ws_et, sum_t, mask, enc_out, out_sum, ws_at, out_cte);

    // 8) proj_d = h @ Wd_s + bd_s                  (M=128, N=256, K=256)
    gemm_bias_kernel<<<dim3(4, 1), 256, 0, stream>>>(
        out_h, HH, HH, nullptr, 0, Wd_s, HH, HH, nullptr, 0, bd_s, ws_projd, HH, HH, HH);

    // 9) ed = tanh(prev_s @ Wd_prev + proj_d) . vd (M=2560, N=256, K=256)
    {
        size_t shmem = (size_t)(128 * (HH + 8) + 64 * 40) * sizeof(__bf16);
        attn_score_kernel<<<(BB * TT) / 128, 256, shmem, stream>>>(
            prev_s, HH, 8, Wd_prev, HH, ws_projd, TT, vd, ws_ed, HH, HH);
    }

    // 10) decoder attention finish: ct_d
    attn_d_finish_kernel<<<BB, 256, 0, stream>>>(ws_ed, prev_s, ws_ctd);

    // 11) p_gen
    pgen_kernel<<<BB, 256, 0, stream>>>(out_cte, ws_ctd, ws_st, ws_x, W_pg, b_pg, ws_pg);

    // 12) zin = concat(h, ct_e_new, ct_d)
    concat_z_kernel<<<(BB * 4 * HH) / 256, 256, 0, stream>>>(out_h, out_cte, ws_ctd, ws_zin);

    // 13) z = zin @ W_V + b_V                      (M=128, N=256, K=1024)
    gemm_bias_kernel<<<dim3(4, 1), 256, 0, stream>>>(
        ws_zin, 4 * HH, 4 * HH, nullptr, 0, W_V, HH, 4 * HH, nullptr, 0, b_V,
        ws_z, HH, HH, 4 * HH);

    // 14) logits = z @ W_V1 + b_V1                 (M=128, N=50000, K=256)
    gemm_bias_kernel<<<dim3((VV + 63) / 64, 1), 256, 0, stream>>>(
        ws_z, HH, HH, nullptr, 0, W_V1, VV, HH, nullptr, 0, b_V1,
        ws_logit, VV, VV, HH);

    // 15) softmax stats
    softmax_stats_kernel<<<BB, 256, 0, stream>>>(ws_logit, ws_rmax, ws_rsum);

    // 16) final = [p_gen * softmax(logits), zeros]
    final_write_kernel<<<dim3((VNO + 255) / 256, BB), 256, 0, stream>>>(
        ws_logit, ws_rmax, ws_rsum, ws_pg, out_final);

    // 17) scatter-add copy mechanism
    scatter_kernel<<<(BB * SS) / 256, 256, 0, stream>>>(evocab, ws_at, ws_pg, out_final);
}
